// NMP_IntNet_67740224192593
// MI455X (gfx1250) — compile-verified, matrix operand order checked by reading, not run on hardware
//
#include <hip/hip_runtime.h>

// ---------------------------------------------------------------------------
// CDNA5 (gfx1250, wave32) fused MLP message-passing network.
// All matmuls via v_wmma_f32_16x16x32_bf16 (f32 accumulate).
// Weights pre-packed once (bf16, transposed, zero-padded) into workspace.
// ---------------------------------------------------------------------------

typedef __attribute__((ext_vector_type(16))) __bf16 v16bf;
typedef __attribute__((ext_vector_type(8)))  float  v8f;

#define THREADS 256   // 8 waves of 32
#define ROWS    64    // rows of the activation tile held in LDS

// ----- LDS partition -----
#define ACT_ELEMS   (ROWS * 256)          // bf16 elems per activation buffer
#define WBUF_ELEMS  (32768)               // bf16 elems (max Npad*Kpad = 256*128)
#define ACT_BYTES   (ACT_ELEMS * 2)
#define WBUF_BYTES  (WBUF_ELEMS * 2)
#define BBUF_BYTES  (256 * 4)
#define OUTF_BYTES  (ROWS * 96 * 4)
#define GSH_BYTES   (64 * 4)
#define SMEM_BYTES  (2 * ACT_BYTES + WBUF_BYTES + BBUF_BYTES + OUTF_BYTES + GSH_BYTES)

// Packed per-MLP weight pointers (device addresses inside d_ws).
struct MlpP {
    const __bf16* Wt[4];   // [Npad x Kpad] bf16, transposed, zero-padded
    const float*  bias[4]; // [Npad] fp32, zero-padded
};

union Frag16 { v16bf v; uint4 q[2]; };

// Per-lane WMMA 16-bit fragment (16 bf16) from a row-major LDS matrix.
// 'kbase' = k0 + half*8. VGPR0..3 <- K=kbase..+7 ; VGPR4..7 <- K=kbase+16..+23.
__device__ __forceinline__ v16bf load_frag(const __bf16* base, int rowStride,
                                           int row, int kbase) {
    const __bf16* p = base + row * rowStride + kbase;
    Frag16 f;
    f.q[0] = *(const uint4*)(p);
    f.q[1] = *(const uint4*)(p + 16);
    return f.v;
}

// Stage a pre-packed weight block + bias from global (workspace) into LDS.
// Pure b128 copy: global_load_b128 -> ds_store_b128.
__device__ __forceinline__ void stage_packed(const __bf16* __restrict__ src, int elems,
                                             const float* __restrict__ bsrc, int Npad,
                                             __bf16* wbuf, float* bbuf, int tid) {
    const uint4* __restrict__ s = (const uint4*)src;
    uint4* d = (uint4*)wbuf;
    const int n16 = elems >> 3;           // 8 bf16 per uint4
    for (int i = tid; i < n16; i += THREADS) d[i] = s[i];
    for (int i = tid; i < Npad; i += THREADS) bbuf[i] = bsrc[i];
}

// Prefetch the next layer's packed weights toward the WGP (global_prefetch_b8).
__device__ __forceinline__ void prefetch_lines(const void* p, int bytes, int tid) {
    const char* c = (const char*)p;
    for (int off = tid * 128; off < bytes; off += THREADS * 128)
        __builtin_prefetch(c + off, 0, 1);
}

__device__ __forceinline__ void store_tile(const v8f& acc, int mt, int col, int half,
                                           float bv, int N, bool relu, bool toF32,
                                           __bf16* outB, float* outF, int strideF) {
#pragma unroll
    for (int r = 0; r < 8; ++r) {
        const int row = mt * 16 + r + half * 8;   // C layout: lanes>=16 hold M+8
        float v = acc[r] + bv;
        if (relu && v < 0.0f) v = 0.0f;
        if (col < N) {
            if (toF32) outF[row * strideF + col] = v;
            else       outB[row * N + col] = (__bf16)v;
        }
    }
}

// One GEMM layer on the 64-row LDS tile: out = act(A[64 x Kpad] * Wt^T + bias).
// 2-wide N register blocking: one A fragment feeds two B fragments/accumulators.
__device__ __forceinline__ void gemm_layer(const __bf16* A, int Kpad,
                                           const __bf16* Wt, const float* bias,
                                           int N, int Npad,
                                           __bf16* outB, float* outF, int strideF,
                                           bool relu, bool toF32, int tid) {
    const int wave = tid >> 5;
    const int lane = tid & 31;
    const int half = lane >> 4;
    const int l15  = lane & 15;
    const int ntiles = Npad >> 4;

    if ((ntiles & 1) == 0) {
        const int T2 = (ROWS / 16) * (ntiles >> 1);
        for (int st = wave; st < T2; st += 8) {
            const int mt  = st & 3;
            const int nt0 = (st >> 2) << 1;
            v8f acc0 = {}, acc1 = {};
            for (int k0 = 0; k0 < Kpad; k0 += 32) {
                const int kb = k0 + half * 8;
                v16bf a  = load_frag(A,  Kpad, mt * 16 + l15, kb);
                v16bf w0 = load_frag(Wt, Kpad, nt0 * 16 + l15, kb);
                v16bf w1 = load_frag(Wt, Kpad, nt0 * 16 + 16 + l15, kb);
                acc0 = __builtin_amdgcn_wmma_f32_16x16x32_bf16(
                           false, a, false, w0, (short)0, acc0, false, false);
                acc1 = __builtin_amdgcn_wmma_f32_16x16x32_bf16(
                           false, a, false, w1, (short)0, acc1, false, false);
            }
            const int c0 = nt0 * 16 + l15;
            store_tile(acc0, mt, c0,      half, bias[c0],      N, relu, toF32, outB, outF, strideF);
            store_tile(acc1, mt, c0 + 16, half, bias[c0 + 16], N, relu, toF32, outB, outF, strideF);
        }
    } else {
        const int T = (ROWS / 16) * ntiles;
        for (int tile = wave; tile < T; tile += 8) {
            const int mt = tile & 3;
            const int nt = tile >> 2;
            v8f acc = {};
            for (int k0 = 0; k0 < Kpad; k0 += 32) {
                const int kb = k0 + half * 8;
                v16bf a = load_frag(A,  Kpad, mt * 16 + l15, kb);
                v16bf w = load_frag(Wt, Kpad, nt * 16 + l15, kb);
                acc = __builtin_amdgcn_wmma_f32_16x16x32_bf16(
                          false, a, false, w, (short)0, acc, false, false);
            }
            const int col = nt * 16 + l15;
            store_tile(acc, mt, col, half, bias[col], N, relu, toF32, outB, outF, strideF);
        }
    }
}

// 4-layer MLP (Kpad0 -> 128 -> 256 -> 128 -> Nlast; ReLU hidden, linear out)
// on the 64-row bf16 tile in actA. Final layer written fp32 into outF.
__device__ void mlp64(const MlpP& p, int Kpad0, int Nlast, int NpadLast,
                      __bf16* actA, __bf16* actB, __bf16* wbuf, float* bbuf,
                      float* outF, int tid) {
    // layer 0: [64 x Kpad0] -> [64 x 128]
    stage_packed(p.Wt[0], 128 * Kpad0, p.bias[0], 128, wbuf, bbuf, tid);
    prefetch_lines(p.Wt[1], 256 * 128 * 2, tid);
    __syncthreads();
    gemm_layer(actA, Kpad0, wbuf, bbuf, 128, 128, actB, nullptr, 0, true, false, tid);
    __syncthreads();
    // layer 1: [64 x 128] -> [64 x 256]
    stage_packed(p.Wt[1], 256 * 128, p.bias[1], 256, wbuf, bbuf, tid);
    prefetch_lines(p.Wt[2], 128 * 256 * 2, tid);
    __syncthreads();
    gemm_layer(actB, 128, wbuf, bbuf, 256, 256, actA, nullptr, 0, true, false, tid);
    __syncthreads();
    // layer 2: [64 x 256] -> [64 x 128]
    stage_packed(p.Wt[2], 128 * 256, p.bias[2], 128, wbuf, bbuf, tid);
    prefetch_lines(p.Wt[3], NpadLast * 128 * 2, tid);
    __syncthreads();
    gemm_layer(actA, 256, wbuf, bbuf, 128, 128, actB, nullptr, 0, true, false, tid);
    __syncthreads();
    // layer 3: [64 x 128] -> [64 x Nlast] fp32, linear
    stage_packed(p.Wt[3], NpadLast * 128, p.bias[3], NpadLast, wbuf, bbuf, tid);
    __syncthreads();
    gemm_layer(actB, 128, wbuf, bbuf, Nlast, NpadLast, nullptr, outF, Nlast, false, true, tid);
    __syncthreads();
}

// ---------------------------------------------------------------------------
// Weight packing: one block per layer. W (K x N fp32) -> Wt (Npad x Kpad bf16,
// transposed, zero-padded); bias -> padded fp32.
// ---------------------------------------------------------------------------
struct PackDesc {
    const float* W; const float* b;
    __bf16* Wt; float* bp;
    int K, Kpad, N, Npad;
};
struct PackArgs { PackDesc d[20]; };

__global__ __launch_bounds__(THREADS)
void pack_kernel(PackArgs a) {
    const PackDesc d = a.d[blockIdx.x];
    const int total = d.Npad * d.Kpad;
    for (int i = threadIdx.x; i < total; i += THREADS) {
        const int n = i / d.Kpad;
        const int k = i - n * d.Kpad;
        const float v = (n < d.N && k < d.K) ? d.W[k * d.N + n] : 0.0f;
        d.Wt[i] = (__bf16)v;
    }
    for (int i = threadIdx.x; i < d.Npad; i += THREADS)
        d.bp[i] = (i < d.N) ? d.b[i] : 0.0f;
}

// ---------------------------------------------------------------------------
// Message kernel: one block per (b, v). Builds 64 pair rows [h_v|h_w|e_vw]
// in LDS, runs the message MLP, gated-sums over w.
// ---------------------------------------------------------------------------
__global__ __launch_bounds__(THREADS)
void msg_kernel(const float* __restrict__ g, const float* __restrict__ h, int d,
                const float* __restrict__ e, MlpP p, int Kpad0,
                float* __restrict__ mOut /* [B*N, 96] */) {
    extern __shared__ char smem[];
    __bf16* actA = (__bf16*)smem;
    __bf16* actB = actA + ACT_ELEMS;
    __bf16* wbuf = actB + ACT_ELEMS;
    float*  bbuf = (float*)((char*)wbuf + WBUF_BYTES);
    float*  outF = (float*)((char*)bbuf + BBUF_BYTES);
    float*  gsh  = (float*)((char*)outF + OUTF_BYTES);

    const int tid = threadIdx.x;
    const int blk = blockIdx.x;           // b*64 + v
    const int b   = blk >> 6;
    const int v   = blk & 63;
    const float* hb = h + (size_t)b * 64 * d;

    for (int i = tid; i < ROWS * Kpad0; i += THREADS) {
        const int w = i / Kpad0;
        const int k = i - w * Kpad0;
        float val = 0.0f;
        if (k < d)               val = hb[v * d + k];
        else if (k < 2 * d)      val = hb[w * d + (k - d)];
        else if (k < 2 * d + 16) val = e[((size_t)blk * 64 + w) * 16 + (k - 2 * d)];
        actA[i] = (__bf16)val;
    }
    if (tid < 64) gsh[tid] = g[(size_t)blk * 64 + tid];
    __syncthreads();

    mlp64(p, Kpad0, 96, 96, actA, actB, wbuf, bbuf, outF, tid);

    for (int c = tid; c < 96; c += THREADS) {
        float s = 0.0f;
#pragma unroll 8
        for (int w = 0; w < 64; ++w) s += gsh[w] * outF[w * 96 + c];
        mOut[(size_t)blk * 96 + c] = s;
    }
}

// ---------------------------------------------------------------------------
// Update kernel: one block per 64 nodes; input = concat(h, m).
// ---------------------------------------------------------------------------
__global__ __launch_bounds__(THREADS)
void upd_kernel(const float* __restrict__ hcur, int d,
                const float* __restrict__ mIn, MlpP p, int Kpad0,
                float* __restrict__ hOut /* [B*N, 96] */) {
    extern __shared__ char smem[];
    __bf16* actA = (__bf16*)smem;
    __bf16* actB = actA + ACT_ELEMS;
    __bf16* wbuf = actB + ACT_ELEMS;
    float*  bbuf = (float*)((char*)wbuf + WBUF_BYTES);
    float*  outF = (float*)((char*)bbuf + BBUF_BYTES);

    const int tid  = threadIdx.x;
    const int base = blockIdx.x * ROWS;

    for (int i = tid; i < ROWS * Kpad0; i += THREADS) {
        const int r = i / Kpad0;
        const int k = i - r * Kpad0;
        const int node = base + r;
        float val = 0.0f;
        if (k < d)           val = hcur[(size_t)node * d + k];
        else if (k < d + 96) val = mIn[(size_t)node * 96 + (k - d)];
        actA[i] = (__bf16)val;
    }
    __syncthreads();

    mlp64(p, Kpad0, 96, 96, actA, actB, wbuf, bbuf, outF, tid);

    for (int i = tid; i < ROWS * 96; i += THREADS)
        hOut[(size_t)base * 96 + i] = outF[i];
}

// ---------------------------------------------------------------------------
// Readout kernel: single block. Sum over nodes, then MLP 96->...->12.
// ---------------------------------------------------------------------------
__global__ __launch_bounds__(THREADS)
void readout_kernel(const float* __restrict__ hIn /* [32*64, 96] */,
                    MlpP p, float* __restrict__ out /* [32, 12] */) {
    extern __shared__ char smem[];
    __bf16* actA = (__bf16*)smem;
    __bf16* actB = actA + ACT_ELEMS;
    __bf16* wbuf = actB + ACT_ELEMS;
    float*  bbuf = (float*)((char*)wbuf + WBUF_BYTES);
    float*  outF = (float*)((char*)bbuf + BBUF_BYTES);

    const int tid = threadIdx.x;

    for (int i = tid; i < ROWS * 96; i += THREADS) {
        const int r = i / 96;
        const int c = i - r * 96;
        float val = 0.0f;
        if (r < 32) {
            const float* hb = hIn + ((size_t)r * 64) * 96 + c;
#pragma unroll 8
            for (int v = 0; v < 64; ++v) val += hb[(size_t)v * 96];
        }
        actA[i] = (__bf16)val;
    }
    __syncthreads();

    mlp64(p, 96, 12, 16, actA, actB, wbuf, bbuf, outF, tid);

    for (int i = tid; i < 32 * 12; i += THREADS) out[i] = outF[i];
}

// ---------------------------------------------------------------------------
// Host launcher
// ---------------------------------------------------------------------------
struct LayerDim { int K, Kpad, N, Npad; };
static const LayerDim LD[5][4] = {
    {{144,160,128,128},{128,128,256,256},{256,256,128,128},{128,128, 96, 96}}, // m0
    {{208,224,128,128},{128,128,256,256},{256,256,128,128},{128,128, 96, 96}}, // m1
    {{160,160,128,128},{128,128,256,256},{256,256,128,128},{128,128, 96, 96}}, // u0
    {{192,192,128,128},{128,128,256,256},{256,256,128,128},{128,128, 96, 96}}, // u1
    {{ 96, 96,128,128},{128,128,256,256},{256,256,128,128},{128,128, 12, 16}}, // rd
};
static const int MLP_IN_BASE[5] = {3, 11, 19, 27, 35};

extern "C" void kernel_launch(void* const* d_in, const int* in_sizes, int n_in,
                              void* d_out, int out_size, void* d_ws, size_t ws_size,
                              hipStream_t stream) {
    (void)in_sizes; (void)n_in; (void)out_size; (void)ws_size;

    const float* g    = (const float*)d_in[0];  // [32,64,64]
    const float* h_in = (const float*)d_in[1];  // [32,64,64]
    const float* e    = (const float*)d_in[2];  // [32,64,64,16]

    // ----- workspace layout -----
    float* mBuf = (float*)d_ws;                 // [2048, 96]
    float* h1   = mBuf + 2048 * 96;             // [2048, 96]
    float* h2   = h1   + 2048 * 96;             // [2048, 96]
    __bf16* wCur = (__bf16*)(h2 + 2048 * 96);   // packed weights (bf16)

    PackArgs pa;
    MlpP mlp[5];
    // First pass: assign Wt segments.
    for (int m2 = 0; m2 < 5; ++m2) {
        for (int l = 0; l < 4; ++l) {
            const LayerDim& ld = LD[m2][l];
            PackDesc& pd = pa.d[m2 * 4 + l];
            pd.W = (const float*)d_in[MLP_IN_BASE[m2] + 2 * l + 0];
            pd.b = (const float*)d_in[MLP_IN_BASE[m2] + 2 * l + 1];
            pd.K = ld.K; pd.Kpad = ld.Kpad; pd.N = ld.N; pd.Npad = ld.Npad;
            pd.Wt = wCur;
            mlp[m2].Wt[l] = wCur;
            wCur += (size_t)ld.Npad * ld.Kpad;
        }
    }
    // Bias segments after all weights (keep 16B alignment).
    float* bCur = (float*)(((uintptr_t)wCur + 15) & ~(uintptr_t)15);
    for (int m2 = 0; m2 < 5; ++m2) {
        for (int l = 0; l < 4; ++l) {
            PackDesc& pd = pa.d[m2 * 4 + l];
            pd.bp = bCur;
            mlp[m2].bias[l] = bCur;
            bCur += pd.Npad;
        }
    }

    hipFuncSetAttribute((const void*)msg_kernel,
                        hipFuncAttributeMaxDynamicSharedMemorySize, SMEM_BYTES);
    hipFuncSetAttribute((const void*)upd_kernel,
                        hipFuncAttributeMaxDynamicSharedMemorySize, SMEM_BYTES);
    hipFuncSetAttribute((const void*)readout_kernel,
                        hipFuncAttributeMaxDynamicSharedMemorySize, SMEM_BYTES);

    // One-time weight packing (cheap: <1MB total).
    pack_kernel<<<20, THREADS, 0, stream>>>(pa);

    // t = 0
    msg_kernel<<<2048, THREADS, SMEM_BYTES, stream>>>(g, h_in, 64, e, mlp[0], 160, mBuf);
    upd_kernel<<<32,   THREADS, SMEM_BYTES, stream>>>(h_in, 64, mBuf, mlp[2], 160, h1);
    // t = 1
    msg_kernel<<<2048, THREADS, SMEM_BYTES, stream>>>(g, h1, 96, e, mlp[1], 224, mBuf);
    upd_kernel<<<32,   THREADS, SMEM_BYTES, stream>>>(h1, 96, mBuf, mlp[3], 192, h2);
    // readout
    readout_kernel<<<1, THREADS, SMEM_BYTES, stream>>>(h2, mlp[4], (float*)d_out);
}
